// MultiHeadFwdAttentionLayer_5987184410674
// MI455X (gfx1250) — matrix operand order repaired and checked by reading, not compiled, not run on hardware
//
#include <hip/hip_runtime.h>
#include <math.h>

#define NNODES 50000
#define NEDGES 800000
#define HIDDEN 128
#define NHEADS 8
#define MLP_IN 336
#define KP     352      // MLP_IN padded to multiple of 32 for f16 WMMA
#define WIDTH  256

#define EPB 64          // edges per block in the MLP kernel
#define XSH 360         // LDS row stride (halves) for the f16 input tile: conflict-free
#define HSTRIDE 260     // LDS row stride (floats) for the 256-wide hidden tile
#define HID_BYTES (EPB * HSTRIDE * 4)   // 66560 > EPB*XSH*2 = 46080 (overlay max)

typedef float    v2f  __attribute__((ext_vector_type(2)));
typedef float    v8f  __attribute__((ext_vector_type(8)));
typedef _Float16 v4h  __attribute__((ext_vector_type(4)));
typedef _Float16 v8h  __attribute__((ext_vector_type(8)));
typedef _Float16 v16h __attribute__((ext_vector_type(16)));

__device__ __forceinline__ v8f wmma16(v16h a, v16h b, v8f c) {
  // D(16x16 f32) = A(16x32 f16) * B(32x16 f16) + C
  return __builtin_amdgcn_wmma_f32_16x16x32_f16(false, a, false, b, (short)0, c, false, false);
}
__device__ __forceinline__ v8f wmma4(v2f a, v2f b, v8f c) {
  // D(16x16 f32) = A(16x4 f32) * B(4x16 f32) + C
  return __builtin_amdgcn_wmma_f32_16x16x4_f32(false, a, false, b, (short)0, c, false, false);
}

// A-fragment: per-lane chunks K[+0..7] and K[+16..23] (caller applies lane-half offset)
__device__ __forceinline__ v16h ldA(const _Float16* p) {
  v8h lo = *(const v8h*)p;
  v8h hi = *(const v8h*)(p + 16);
  return __builtin_shufflevector(lo, hi, 0,1,2,3,4,5,6,7,8,9,10,11,12,13,14,15);
}

// order-monotone float <-> uint encoding for atomicMax-based segment max
__device__ __forceinline__ unsigned encF(float f) {
  unsigned u = __float_as_uint(f);
  return (u & 0x80000000u) ? ~u : (u | 0x80000000u);
}
__device__ __forceinline__ float decF(unsigned u) {
  return __uint_as_float((u & 0x80000000u) ? (u & 0x7fffffffu) : ~u);
}

// ---------------- kernel 0: init workspace ----------------
__global__ void k0_init(unsigned* __restrict__ maxenc, float* __restrict__ nsum,
                        float* __restrict__ agg) {
  int i = blockIdx.x * blockDim.x + threadIdx.x;
  if (i < NNODES * NHEADS) { maxenc[i] = 0x007fffffu; /* enc(-inf) */ nsum[i] = 0.0f; }
  if (i < NNODES * HIDDEN) agg[i] = 0.0f;
}

// ---------------- kernel 0b: W1 fp32 -> f16, K padded to 352 ----------------
__global__ void k0b_cvt_w1(const float* __restrict__ W1, _Float16* __restrict__ W1h) {
  int i = blockIdx.x * blockDim.x + threadIdx.x;
  if (i >= WIDTH * KP) return;
  int n = i / KP, k = i % KP;
  W1h[i] = (k < MLP_IN) ? (_Float16)W1[n * MLP_IN + k] : (_Float16)0.0f;
}

// ---------------- kernel 1: edge MLP (f16 WMMA) + scores + segment max ----------------
__global__ __launch_bounds__(256) void k1_edge_mlp(
    const float* __restrict__ h, const float* __restrict__ xs_in,
    const int* __restrict__ src, const int* __restrict__ dst,
    const float* __restrict__ ef, const _Float16* __restrict__ W1h,
    const float* __restrict__ W2, float* __restrict__ scores,
    unsigned* __restrict__ maxenc)
{
  extern __shared__ char smemraw[];
  _Float16* xh  = (_Float16*)smemraw;             // [EPB][XSH] f16 input tile
  float*    hid = (float*)smemraw;                // overlay: [EPB][HSTRIDE] f32 hidden
  int* s_src = (int*)(smemraw + HID_BYTES);
  int* s_dst = s_src + EPB;

  const int tid  = threadIdx.x;
  const int lane = tid & 31;
  const int wv   = tid >> 5;
  const int e0   = blockIdx.x * EPB;

  if (tid < EPB) { s_src[tid] = src[e0 + tid]; s_dst[tid] = dst[e0 + tid]; }
  __syncthreads();

  // Stage X tile [64 x 352] f16 = concat(h[src], h[dest], x_s[src], x_s[dest], ef, 0-pad)
  for (int idx = tid; idx < EPB * 88; idx += 256) {
    int e = idx / 88, c = idx % 88;               // c = 4-element chunk within row
    v4h o;
    if (c < 84) {
      float4 v;
      if (c < 32)       v = ((const float4*)h)[(size_t)s_src[e] * 32 + c];
      else if (c < 64)  v = ((const float4*)h)[(size_t)s_dst[e] * 32 + (c - 32)];
      else if (c < 72)  v = ((const float4*)xs_in)[(size_t)s_src[e] * 8 + (c - 64)];
      else if (c < 80)  v = ((const float4*)xs_in)[(size_t)s_dst[e] * 8 + (c - 72)];
      else              v = ((const float4*)ef)[(size_t)(e0 + e) * 4 + (c - 80)];
      o = (v4h){(_Float16)v.x, (_Float16)v.y, (_Float16)v.z, (_Float16)v.w};
    } else {
      o = (v4h){(_Float16)0.f, (_Float16)0.f, (_Float16)0.f, (_Float16)0.f};
    }
    *(v4h*)&xh[e * XSH + c * 4] = o;
  }
  __syncthreads();

  // hid[64x256] = relu(X @ W1^T) via f16 WMMA, f32 accumulate.
  // Wave wv owns output cols [32*wv, 32*wv+32).
  const int m     = lane & 15;
  const int koffA = (lane >> 4) << 3;             // A: +0 / +8 per lane half
  const int koffB = (lane >> 4) << 3;             // B: +0 / +8 per lane half
  const int col0  = wv * 32 + m;
  const int col1  = col0 + 16;
  const v8f vz = {0.f,0.f,0.f,0.f,0.f,0.f,0.f,0.f};
  v8f acc0[4], acc1[4];
  #pragma unroll
  for (int t = 0; t < 4; ++t) { acc0[t] = vz; acc1[t] = vz; }

  for (int k0 = 0; k0 < KP; k0 += 32) {
    v16h b0 = ldA(&W1h[col0 * KP + k0 + koffB]);
    v16h b1 = ldA(&W1h[col1 * KP + k0 + koffB]);
    #pragma unroll
    for (int t = 0; t < 4; ++t) {
      v16h a = ldA(&xh[(t * 16 + m) * XSH + k0 + koffA]);
      acc0[t] = wmma16(a, b0, acc0[t]);
      acc1[t] = wmma16(a, b1, acc1[t]);
    }
  }
  __syncthreads();                                // everyone done reading X

  // ReLU + park hid in LDS (f32, row stride 260 avoids bank conflicts)
  #pragma unroll
  for (int t = 0; t < 4; ++t) {
    #pragma unroll
    for (int r = 0; r < 8; ++r) {
      int row = t * 16 + ((lane < 16) ? r : (8 + r));
      hid[row * HSTRIDE + col0] = fmaxf(acc0[t][r], 0.0f);
      hid[row * HSTRIDE + col1] = fmaxf(acc1[t][r], 0.0f);
    }
  }
  __syncthreads();

  // scores[64x8] = leaky_relu(hid @ W2^T) / 4 ; segment max via atomicMax (f32 exact)
  for (int o = tid; o < EPB * NHEADS; o += 256) {
    int e = o >> 3, hd = o & 7;
    float s = 0.0f;
    #pragma unroll 8
    for (int k = 0; k < WIDTH; ++k)
      s += hid[e * HSTRIDE + k] * W2[hd * WIDTH + k];
    s = (s > 0.0f) ? s : 0.01f * s;               // leaky_relu, slope 0.01
    s *= 0.25f;                                   // / sqrt(HEAD_SIZE=16)
    scores[(size_t)(e0 + e) * NHEADS + hd] = s;
    atomicMax(&maxenc[s_dst[e] * NHEADS + hd], encF(s));
  }
}

// ---------------- kernel 2: exp(score - segmax), segment sum ----------------
__global__ void k2_exp(const int* __restrict__ dst, float* __restrict__ scores,
                       const unsigned* __restrict__ maxenc, float* __restrict__ nsum) {
  int i = blockIdx.x * blockDim.x + threadIdx.x;
  if (i >= NEDGES * NHEADS) return;
  int e = i >> 3, hd = i & 7;
  int d = dst[e];
  float w = expf(scores[i] - decF(maxenc[d * NHEADS + hd]));
  scores[i] = w;                                  // in-place: now holds exp weights
  atomicAdd(&nsum[d * NHEADS + hd], w);
}

// ---------------- kernel 3: normalize + weighted scatter-add ----------------
__global__ __launch_bounds__(256) void k3_weight_agg(
    const float* __restrict__ h, const int* __restrict__ src,
    const int* __restrict__ dst, const float* __restrict__ expw,
    const float* __restrict__ nsum, float* __restrict__ agg,
    float* __restrict__ out_w)
{
  const int lane = threadIdx.x & 31;
  const int e = blockIdx.x * 8 + (threadIdx.x >> 5);   // one wave per edge
  const int s = src[e], d = dst[e];
  if (lane < NHEADS) {
    float w = expw[(size_t)e * NHEADS + lane] / (nsum[d * NHEADS + lane] + 1e-9f);
    out_w[(size_t)e * NHEADS + lane] = w;
  }
  const int hd = lane >> 2;                            // 4 floats/lane => 1 head / 4 lanes
  float w = expw[(size_t)e * NHEADS + hd] / (nsum[d * NHEADS + hd] + 1e-9f);
  float4 hv = ((const float4*)h)[(size_t)s * 32 + lane];
  float* ap = agg + (size_t)d * HIDDEN + lane * 4;
  atomicAdd(ap + 0, hv.x * w);
  atomicAdd(ap + 1, hv.y * w);
  atomicAdd(ap + 2, hv.z * w);
  atomicAdd(ap + 3, hv.w * w);
}

// ---------------- kernel 4: projected = agg @ W_out^T (exact fp32 WMMA) ----------------
__global__ __launch_bounds__(256) void k4_project(
    const float* __restrict__ agg, const float* __restrict__ Wout,
    float* __restrict__ out)
{
  const int lane  = threadIdx.x & 31;
  const int wv    = threadIdx.x >> 5;                  // wave -> col tile (8 tiles = 128)
  const int r0    = blockIdx.x * 16;
  const int m     = lane & 15;
  const int khalf = (lane >> 4) << 1;
  const int col   = wv * 16 + m;
  v8f acc = {0.f,0.f,0.f,0.f,0.f,0.f,0.f,0.f};
  #pragma unroll
  for (int k0 = 0; k0 < HIDDEN; k0 += 4) {
    const int kk = k0 + khalf;
    v2f a = *(const v2f*)&agg[(size_t)(r0 + m) * HIDDEN + kk];
    v2f b = *(const v2f*)&Wout[col * HIDDEN + kk];
    acc = wmma4(a, b, acc);
  }
  #pragma unroll
  for (int r = 0; r < 8; ++r) {
    int row = r0 + ((lane < 16) ? r : (8 + r));
    out[(size_t)row * HIDDEN + col] = acc[r];
  }
}

// ---------------- host launch ----------------
extern "C" void kernel_launch(void* const* d_in, const int* in_sizes, int n_in,
                              void* d_out, int out_size, void* d_ws, size_t ws_size,
                              hipStream_t stream) {
  const float* h  = (const float*)d_in[0];
  const float* xs = (const float*)d_in[1];
  const int*   sr = (const int*)d_in[2];
  const int*   ds = (const int*)d_in[3];
  const float* ef = (const float*)d_in[4];
  const float* W1 = (const float*)d_in[5];
  const float* W2 = (const float*)d_in[6];
  const float* Wo = (const float*)d_in[7];

  float* out_proj = (float*)d_out;                     // [N, 128]
  float* out_w    = out_proj + (size_t)NNODES * HIDDEN;// [E, 8]

  // ws layout (floats): scores/expw [E*8] | maxenc [N*8] | nsum [N*8] | agg [N*128] | W1h (f16)
  float*    scores = (float*)d_ws;
  unsigned* maxenc = (unsigned*)(scores + (size_t)NEDGES * NHEADS);
  float*    nsum   = (float*)(maxenc + (size_t)NNODES * NHEADS);
  float*    agg    = nsum + (size_t)NNODES * NHEADS;
  _Float16* W1h    = (_Float16*)(agg + (size_t)NNODES * HIDDEN);

  k0_init<<<(NNODES * HIDDEN + 255) / 256, 256, 0, stream>>>(maxenc, nsum, agg);
  k0b_cvt_w1<<<(WIDTH * KP + 255) / 256, 256, 0, stream>>>(W1, W1h);

  const size_t shbytes = HID_BYTES + 2 * EPB * sizeof(int);
  k1_edge_mlp<<<NEDGES / EPB, 256, shbytes, stream>>>(h, xs, sr, ds, ef, W1h, W2,
                                                      scores, maxenc);
  k2_exp<<<(NEDGES * NHEADS + 255) / 256, 256, 0, stream>>>(ds, scores, maxenc, nsum);
  k3_weight_agg<<<NEDGES / 8, 256, 0, stream>>>(h, sr, ds, scores, nsum, agg, out_w);
  k4_project<<<NNODES / 16, 256, 0, stream>>>(agg, Wo, out_proj);
}
